// Nystroem_75273596830261
// MI455X (gfx1250) — matrix operand order repaired and checked by reading, not compiled, not run on hardware
//
#include <hip/hip_runtime.h>
#include <cstdint>

typedef __bf16 bf16;
typedef __attribute__((ext_vector_type(16))) __bf16 v16bf;
typedef __attribute__((ext_vector_type(8)))  __bf16 v8bf;
typedef __attribute__((ext_vector_type(8)))  float  v8f;

#define NROWS 16384
#define MCENT 1024
#define DDIM  1024

// ---------------------------------------------------------------------------
// Row-wise squared-norm + f32 -> bf16 conversion. One wave32 per row.
// ---------------------------------------------------------------------------
__global__ __launch_bounds__(256) void rownorm_convert_k(
    const float* __restrict__ src, bf16* __restrict__ dst,
    float* __restrict__ sq, int cols)
{
  const int wave = threadIdx.x >> 5;
  const int lane = threadIdx.x & 31;
  const long row = (long)blockIdx.x * 8 + wave;
  const float* s = src + row * (long)cols;
  bf16* d = dst + row * (long)cols;
  float acc = 0.f;
  for (int c = lane; c < cols; c += 32) {
    float v = s[c];
    acc += v * v;
    d[c] = (bf16)v;
  }
  for (int off = 16; off; off >>= 1) acc += __shfl_xor(acc, off, 32);
  if (lane == 0) sq[row] = acc;
}

// ---------------------------------------------------------------------------
// Transpose + convert Norm (MxM f32, row-major) -> bf16 transposed.
// ---------------------------------------------------------------------------
__global__ __launch_bounds__(256) void transpose_convert_k(
    const float* __restrict__ src, bf16* __restrict__ dst, int m)
{
  long idx = (long)blockIdx.x * blockDim.x + threadIdx.x;
  int n = (int)(idx / m);
  int k = (int)(idx % m);
  dst[(long)n * m + k] = (bf16)src[(long)k * m + n];
}

// ---------------------------------------------------------------------------
// bf16 WMMA GEMM with async global->LDS double-buffered pipeline.
//   acc[r,c] = sum_k A[r,k] * BT[c,k]   (f32 accumulate)
//   EXP epilogue:  outH[r,c] = bf16(exp(-g*(xsq[r]+csq[c]-2*acc)))
//   else:          outF[r,c] = acc
// ---------------------------------------------------------------------------
#define BM 128
#define BN 128
#define BKK 32
#define LDT 40   // padded LDS row stride in bf16 elems (80B) to spread banks

template<bool EXP>
__global__ __launch_bounds__(256) void gemm_wmma_k(
    const bf16* __restrict__ A,    // [rows x K] row-major
    const bf16* __restrict__ BT,   // [cols x K] row-major (B transposed)
    int K,
    float* __restrict__ outF, bf16* __restrict__ outH, int ldc,
    const float* __restrict__ xsq, const float* __restrict__ csq,
    const float* __restrict__ gptr)
{
  __shared__ bf16 As[2][BM * LDT];
  __shared__ bf16 Bs[2][BN * LDT];

  const int t     = threadIdx.x;
  const int wave  = t >> 5;
  const int lane  = t & 31;
  const int wm    = wave >> 2;        // 0..1  (row dir, 64 rows each)
  const int wn    = wave & 3;         // 0..3  (col dir, 32 cols each)
  const int lmod  = lane & 15;
  const int lhalf = lane >> 4;
  const long rowBlock = (long)blockIdx.y * BM;
  const long colBlock = (long)blockIdx.x * BN;

  // staging: thread t copies 32B (2 x b128 async) of row (t>>1), seg (t&1)
  const int  srow = t >> 1;
  const int  sseg = (t & 1) * 16;
  const bf16* gA = A  + (rowBlock + srow) * (long)K + sseg;
  const bf16* gB = BT + (colBlock + srow) * (long)K + sseg;

  // Async copy one 32x128 K-tile of A and B into LDS buffer `buf`.
  // INST_OFFSET adds to both LDS and global addresses (ISA 08 §4.4), so the
  // second 16B chunk is the same registers with offset:16.
  auto issue_tile = [&](int k0, int buf) {
    unsigned la = (unsigned)(uintptr_t)&As[buf][srow * LDT + sseg];
    unsigned lb = (unsigned)(uintptr_t)&Bs[buf][srow * LDT + sseg];
    const bf16* pa = gA + k0;
    const bf16* pb = gB + k0;
    asm volatile(
        "global_load_async_to_lds_b128 %0, %2, off\n\t"
        "global_load_async_to_lds_b128 %0, %2, off offset:16\n\t"
        "global_load_async_to_lds_b128 %1, %3, off\n\t"
        "global_load_async_to_lds_b128 %1, %3, off offset:16"
        :: "v"(la), "v"(lb), "v"(pa), "v"(pb)
        : "memory");
  };

  v8f acc[4][2] = {};

  issue_tile(0, 0);

  int buf = 0;
  for (int k0 = 0; k0 < K; k0 += BKK, buf ^= 1) {
    // My async writes done; barrier makes ALL waves' writes to `buf` visible
    // and guarantees everyone finished reading buf^1 (previous compute).
    asm volatile("s_wait_asynccnt 0x0" ::: "memory");
    __syncthreads();
    if (k0 + BKK < K) issue_tile(k0 + BKK, buf ^ 1);

    // --- build WMMA fragments per ISA 7.12.2 (wave32) ---
    v16bf af[4], bfv[2];
#pragma unroll
    for (int mt = 0; mt < 4; ++mt) {
      // A 16x32 bf16: lane = row M (L%16); elems 0-7 = K[8h..8h+7],
      // elems 8-15 = K[16+8h..16+8h+7], h = L/16
      const bf16* p = &As[buf][(wm * 64 + mt * 16 + lmod) * LDT + 8 * lhalf];
      v8bf lo = *(const v8bf*)p;
      v8bf hi = *(const v8bf*)(p + 16);
      af[mt] = __builtin_shufflevector(lo, hi,
               0,1,2,3,4,5,6,7,8,9,10,11,12,13,14,15);
    }
#pragma unroll
    for (int nt = 0; nt < 2; ++nt) {
      // B 32x16 bf16: lane = col N (L%16); elems 0-7 = K[8h..8h+7],
      // elems 8-15 = K[16+8h..16+8h+7], h = L/16
      const bf16* p = &Bs[buf][(wn * 32 + nt * 16 + lmod) * LDT + 8 * lhalf];
      v8bf lo = *(const v8bf*)p;
      v8bf hi = *(const v8bf*)(p + 16);
      bfv[nt] = __builtin_shufflevector(lo, hi,
               0,1,2,3,4,5,6,7,8,9,10,11,12,13,14,15);
    }

#pragma unroll
    for (int mt = 0; mt < 4; ++mt)
#pragma unroll
      for (int nt = 0; nt < 2; ++nt)
        acc[mt][nt] = __builtin_amdgcn_wmma_f32_16x16x32_bf16(
            false, af[mt], false, bfv[nt], (short)0, acc[mt][nt],
            false, false);
  }

  // --- epilogue; C/D layout: elem i -> row i + 8*(L/16), col L%16 ---
  const float g = EXP ? gptr[0] : 0.f;
#pragma unroll
  for (int mt = 0; mt < 4; ++mt) {
#pragma unroll
    for (int nt = 0; nt < 2; ++nt) {
      const long col = colBlock + wn * 32 + nt * 16 + lmod;
      const float cs = EXP ? csq[col] : 0.f;
#pragma unroll
      for (int i = 0; i < 8; ++i) {
        const long row = rowBlock + wm * 64 + mt * 16 + i + 8 * lhalf;
        const float v = acc[mt][nt][i];
        if (EXP) {
          const float dist = xsq[row] + cs - 2.0f * v;
          outH[row * (long)ldc + col] = (bf16)__expf(-g * dist);
        } else {
          outF[row * (long)ldc + col] = v;
        }
      }
    }
  }
}

// ---------------------------------------------------------------------------
extern "C" void kernel_launch(void* const* d_in, const int* in_sizes, int n_in,
                              void* d_out, int out_size, void* d_ws, size_t ws_size,
                              hipStream_t stream)
{
  const float* X     = (const float*)d_in[0];  // [N, D]
  const float* C     = (const float*)d_in[1];  // [M, D]
  const float* gamma = (const float*)d_in[2];  // [1]
  const float* Norm  = (const float*)d_in[3];  // [M, M]
  float* out = (float*)d_out;                  // [N, M]

  const int N = NROWS, M = MCENT, D = DDIM;

  char* w = (char*)d_ws;
  auto carve = [&](size_t bytes) -> char* {
    char* p = w;
    w += (bytes + 255) & ~(size_t)255;
    return p;
  };
  bf16*  Xb  = (bf16*)carve((size_t)N * D * sizeof(bf16));   // 32 MB
  bf16*  Cb  = (bf16*)carve((size_t)M * D * sizeof(bf16));   //  2 MB
  bf16*  NbT = (bf16*)carve((size_t)M * M * sizeof(bf16));   //  2 MB
  bf16*  Db  = (bf16*)carve((size_t)N * M * sizeof(bf16));   // 32 MB
  float* xsq = (float*)carve((size_t)N * sizeof(float));
  float* csq = (float*)carve((size_t)M * sizeof(float));

  rownorm_convert_k<<<N / 8, 256, 0, stream>>>(X, Xb, xsq, D);
  rownorm_convert_k<<<M / 8, 256, 0, stream>>>(C, Cb, csq, D);
  transpose_convert_k<<<(int)(((long)M * M) / 256), 256, 0, stream>>>(Norm, NbT, M);

  // Phase 1: cross = Xb @ Cb^T, fused Gaussian epilogue -> Db (bf16)
  gemm_wmma_k<true><<<dim3(M / BN, N / BM), 256, 0, stream>>>(
      Xb, Cb, D, nullptr, Db, M, xsq, csq, gamma);

  // Phase 2: out = Db @ Norm   (B^T = NbT), f32 output
  gemm_wmma_k<false><<<dim3(M / BN, N / BM), 256, 0, stream>>>(
      Db, NbT, M, out, nullptr, M, nullptr, nullptr, nullptr);
}
